// EntityModel_41489384079759
// MI455X (gfx1250) — matrix operand order repaired and checked by reading, not compiled, not run on hardware
//
#include <hip/hip_runtime.h>
#include <hip/hip_bf16.h>
#include <math.h>

typedef __attribute__((ext_vector_type(16))) __bf16 v16bf;
typedef __attribute__((ext_vector_type(8)))  __bf16 v8bf;
typedef __attribute__((ext_vector_type(8)))  float  v8f;

#define B_   32
#define NS_  4096
#define S_   512
#define H_   768
#define WE_  150
#define L_   9
#define ROWS (B_*NS_)            /* 131072 */
#define K1   1686                /* 2H+WE  */
#define NC1  53                  /* ceil(1686/32) K-chunks, layer1 */
#define NJ1  10                  /* 160/16 N-tiles, layer1 */
#define NC2  5                   /* 160/32 K-chunks, layer2/3 */
#define NJ2  10
#define NC3  5
#define FRAG_BYTES 1024          /* 32 lanes * 32B */
#define CHUNK_BYTES (NJ1*FRAG_BYTES)   /* 10240B of B-fragments per K-chunk */

/* workspace layout (bytes) */
#define W1_OFF   0
#define W1_BYTES (NC1*NJ1*FRAG_BYTES)          /* 542720 */
#define W2_OFF   (W1_OFF + W1_BYTES)
#define W2_BYTES (NC2*NJ2*FRAG_BYTES)          /* 51200 */
#define W3_OFF   (W2_OFF + W2_BYTES)
#define W3_BYTES (NC3*1*FRAG_BYTES)            /* 5120 */
#define WTB_OFF  (W3_OFF + W3_BYTES)           /* 599040 */
#define WTB_BYTES (9*160*2)                    /* 2880, relu'd, zero-padded */
#define SEQB_OFF (WTB_OFF + WTB_BYTES)         /* 601920, 16B aligned */
#define SEQ_ELEMS ((size_t)B_*S_*H_)           /* 12582912 */

#define ND1 (NC1*NJ1*32*8)
#define ND2 (NC2*NJ2*32*8)
#define ND3 (NC3*1*32*8)
#define NDTOT (ND1+ND2+ND3)
#define WTB_DW (9*160/2)                       /* 720 dwords */

__device__ inline unsigned short bf16_bits(float f) {
  __bf16 b = (__bf16)f;
  return __builtin_bit_cast(unsigned short, b);
}

// ---------------------------------------------------------------------------
// Prep A: relu(sequence_output) -> bf16 copy in workspace.
// ---------------------------------------------------------------------------
__global__ __launch_bounds__(256)
void prep_seq(const float* __restrict__ seq, unsigned* __restrict__ seqb_u) {
  size_t t = (size_t)blockIdx.x * 256 + threadIdx.x;      /* 4 elems/thread */
  if (t >= SEQ_ELEMS / 4) return;
  float4 x = ((const float4*)seq)[t];
  unsigned lo = ((unsigned)bf16_bits(fmaxf(x.y, 0.f)) << 16) | bf16_bits(fmaxf(x.x, 0.f));
  unsigned hi = ((unsigned)bf16_bits(fmaxf(x.w, 0.f)) << 16) | bf16_bits(fmaxf(x.z, 0.f));
  seqb_u[t*2]   = lo;
  seqb_u[t*2+1] = hi;
}

// ---------------------------------------------------------------------------
// Prep B: swizzle W1/W2/W3 into bf16 WMMA B-fragments (32B contiguous per
// lane); build relu'd zero-padded 9x160 bf16 width table; zero loss slot.
// ---------------------------------------------------------------------------
__global__ __launch_bounds__(256)
void prep_swizzle(const float* __restrict__ W1,
                  const float* __restrict__ W2,
                  const float* __restrict__ W3,
                  const float* __restrict__ wtab,
                  unsigned* __restrict__ ws_u,
                  float* __restrict__ loss_slot) {
  int t = blockIdx.x * blockDim.x + threadIdx.x;
  if (t == 0) *loss_slot = 0.0f;

  if (t >= NDTOT) {
    int d = t - NDTOT;
    if (d < WTB_DW) {            /* width table: relu + pad to 160 cols */
      int row = d / 80;
      int cp  = (d % 80) * 2;
      float a0 = (cp     < WE_) ? fmaxf(wtab[row*WE_ + cp],     0.f) : 0.f;
      float a1 = (cp + 1 < WE_) ? fmaxf(wtab[row*WE_ + cp + 1], 0.f) : 0.f;
      ws_u[WTB_OFF/4 + d] = ((unsigned)bf16_bits(a1) << 16) | bf16_bits(a0);
    }
    return;
  }

  const float* W; int K, N, NJ, base_dw, d;
  if (t < ND1)            { W = W1; K = K1;  N = WE_; NJ = NJ1; base_dw = W1_OFF/4; d = t; }
  else if (t < ND1 + ND2) { W = W2; K = WE_; N = WE_; NJ = NJ2; base_dw = W2_OFF/4; d = t - ND1; }
  else                    { W = W3; K = WE_; N = L_;  NJ = 1;   base_dw = W3_OFF/4; d = t - ND1 - ND2; }

  int v = d & 7;
  int l = (d >> 3) & 31;
  int f = d >> 8;
  int c = f / NJ, j = f % NJ;
  int k0 = c*32 + ((l < 16) ? 0 : 8) + 2*v + ((v >= 4) ? 8 : 0);
  int n  = j*16 + (l & 15);

  float v0 = (k0     < K && n < N) ? W[(size_t)k0     * N + n] : 0.0f;
  float v1 = (k0 + 1 < K && n < N) ? W[(size_t)(k0+1) * N + n] : 0.0f;
  ws_u[base_dw + d] = ((unsigned)bf16_bits(v1) << 16) | (unsigned)bf16_bits(v0);
}

// ---------------------------------------------------------------------------
// Fused 3-layer MLP. 8 waves/block, one 16-row span tile per wave.
// ALL weight fragments (W1, W2, W3) are streamed into LDS once per block via
// global_load_async_to_lds_b128 (double buffered, ASYNCcnt-synchronized);
// A and B fragments are software-pipelined one step ahead of each WMMA.
// ---------------------------------------------------------------------------
__global__ __launch_bounds__(256)
void span_mlp_wmma(const float* __restrict__ b1,
                   const float* __restrict__ b2,
                   const float* __restrict__ b3,
                   const int*   __restrict__ spans,
                   const char*  __restrict__ ws,
                   float* __restrict__ out) {
  __shared__ __align__(32) unsigned char bstage[2][CHUNK_BYTES];  /* 20 KB */
  __shared__ __bf16 lds_h[8][16][160];                            /* 40 KB */

  const int lane = threadIdx.x & 31;
  const int wave = threadIdx.x >> 5;
  const int row0 = (blockIdx.x * 8 + wave) * 16;
  const int nloc = lane & 15;
  const int hi   = lane >> 4;            /* which K-half of the A fragment */
  const int moff = hi * 8;               /* D-layout row offset */

  /* span gather setup: lane owns A-row (row0 + nloc) */
  const int r = row0 + nloc;
  const int b = r >> 12;
  const int s_idx = spans[r*3 + 0];
  const int e_idx = spans[r*3 + 1];
  const int w_idx = spans[r*3 + 2];

  const __bf16* seqb = (const __bf16*)(ws + SEQB_OFF);
  const __bf16* ss   = seqb + (size_t)(b*S_ + s_idx) * H_;
  const __bf16* se   = seqb + (size_t)(b*S_ + e_idx) * H_;
  const __bf16* wtr  = (const __bf16*)(ws + WTB_OFF) + (size_t)w_idx * 160;

  const unsigned  bst_base = (unsigned)(uintptr_t)&bstage[0][0];
  const unsigned long long w1_base = (unsigned long long)(uintptr_t)(ws + W1_OFF);
  const unsigned long long w2_base = (unsigned long long)(uintptr_t)(ws + W2_OFF);
  const unsigned long long w3_base = (unsigned long long)(uintptr_t)(ws + W3_OFF);

  /* async-stage `per*5` 512B pieces from gbase into bstage[buf];
     waves 0..4 issue `per` ops apiece (per=4 -> 10KB, per=2 -> 5KB). */
  auto issue_stage = [&](unsigned long long gbase, int buf, int per) {
    if (wave < 5) {
#pragma unroll 4
      for (int q = 0; q < per; ++q) {
        int i = wave*per + q;
        unsigned lds_off = bst_base + (unsigned)buf*CHUNK_BYTES
                         + (unsigned)i*512u + (unsigned)lane*16u;
        unsigned long long ga = gbase + (unsigned long long)(i*512 + lane*16);
        asm volatile("global_load_async_to_lds_b128 %0, %1, off"
                     :: "v"(lds_off), "v"(ga) : "memory");
      }
    }
  };

  /* layer-1 A fragment for K-chunk c: two 16B bf16 runs (already relu'd) */
  auto loadA1 = [&](int c) -> v16bf {
    int k0 = c*32 + hi*8;
    const __bf16* baseA = (c < 24) ? ss : (c < 48) ? (se - H_) : (wtr - 2*H_);
    v8bf r0 = *(const v8bf*)(baseA + k0);
    v8bf r1 = *(const v8bf*)(baseA + k0 + 16);
    v16bf a;
#pragma unroll
    for (int i = 0; i < 8; i++) { a[i] = r0[i]; a[8+i] = r1[i]; }
    return a;
  };

  /* layer-2/3 A fragment from lds_h */
  auto loadA_lds = [&](int c) -> v16bf {
    int kbase = c*32 + hi*8;
    v16bf a;
#pragma unroll
    for (int i = 0; i < 8; i++) a[i]     = lds_h[wave][nloc][kbase + i];
#pragma unroll
    for (int i = 0; i < 8; i++) a[8 + i] = lds_h[wave][nloc][kbase + 16 + i];
    return a;
  };

  const v8f vzero = {0.f,0.f,0.f,0.f,0.f,0.f,0.f,0.f};

  /* ---------------- layer 1: relu(rep) @ W1 + b1, relu ---------------- */
  v8f acc[NJ1];
#pragma unroll
  for (int j = 0; j < NJ1; j++) acc[j] = vzero;

  issue_stage(w1_base, 0, 4);
  v16bf aCur = loadA1(0);
  for (int c = 0; c < NC1; ++c) {
    const int cur = c & 1;
    v16bf aNext;
    if (c + 1 < NC1) {
      issue_stage(w1_base + (unsigned long long)(c+1)*CHUNK_BYTES, cur ^ 1, 4);
      aNext = loadA1(c + 1);
      asm volatile("s_wait_asynccnt 0x4" ::: "memory");   /* chunk c staged */
    } else {
      aNext = aCur;
      asm volatile("s_wait_asynccnt 0x0" ::: "memory");
    }
    __syncthreads();                                      /* publish to all */

    const unsigned char* bst = &bstage[cur][0];
    v16bf bmCur = *(const v16bf*)(bst + lane*32);
#pragma unroll
    for (int j = 0; j < NJ1; j++) {
      v16bf bmNext = (j + 1 < NJ1)
                   ? *(const v16bf*)(bst + (j+1)*FRAG_BYTES + lane*32) : bmCur;
      acc[j] = __builtin_amdgcn_wmma_f32_16x16x32_bf16(
                   false, aCur, false, bmCur, (short)0, acc[j], false, false);
      bmCur = bmNext;
    }
    __syncthreads();                     /* done reading before re-stage */
    aCur = aNext;
  }

  issue_stage(w2_base, 0, 4);            /* prefetch W2 chunk 0 (buf0 free) */

#pragma unroll
  for (int j = 0; j < NJ1; j++) {
    int ng = j*16 + nloc;
    float bias = (ng < WE_) ? b1[ng] : 0.0f;
#pragma unroll
    for (int v = 0; v < 8; v++) {
      float f = fmaxf(acc[j][v] + bias, 0.0f);
      lds_h[wave][moff + v][ng] = (__bf16)f;
    }
  }

  /* ---------------- layer 2: h1 @ W2 + b2 ---------------- */
  v8f acc2[NJ2];
#pragma unroll
  for (int j = 0; j < NJ2; j++) acc2[j] = vzero;

  for (int c = 0; c < NC2; ++c) {
    const int cur = c & 1;
    if (c + 1 < NC2) {
      issue_stage(w2_base + (unsigned long long)(c+1)*CHUNK_BYTES, cur ^ 1, 4);
      asm volatile("s_wait_asynccnt 0x4" ::: "memory");
    } else {
      asm volatile("s_wait_asynccnt 0x0" ::: "memory");
    }
    __syncthreads();                     /* covers lds_h stores + stage c */

    v16bf a = loadA_lds(c);
    const unsigned char* bst = &bstage[cur][0];
    v16bf bmCur = *(const v16bf*)(bst + lane*32);
#pragma unroll
    for (int j = 0; j < NJ2; j++) {
      v16bf bmNext = (j + 1 < NJ2)
                   ? *(const v16bf*)(bst + (j+1)*FRAG_BYTES + lane*32) : bmCur;
      acc2[j] = __builtin_amdgcn_wmma_f32_16x16x32_bf16(
                    false, a, false, bmCur, (short)0, acc2[j], false, false);
      bmCur = bmNext;
    }
    __syncthreads();
  }

  issue_stage(w3_base, 1, 2);            /* stage all of W3 (5KB, buf1 free) */

#pragma unroll
  for (int j = 0; j < NJ2; j++) {
    int ng = j*16 + nloc;
    float bias = (ng < WE_) ? b2[ng] : 0.0f;
#pragma unroll
    for (int v = 0; v < 8; v++) {
      lds_h[wave][moff + v][ng] = (__bf16)(acc2[j][v] + bias);
    }
  }
  asm volatile("s_wait_asynccnt 0x0" ::: "memory");
  __syncthreads();                       /* h2 + W3 fragments visible */

  /* ---------------- layer 3: h2 @ W3 + b3 -> logits ---------------- */
  v8f acc3 = vzero;
  {
    const unsigned char* bst = &bstage[1][0];
    v16bf bmCur = *(const v16bf*)(bst + lane*32);
#pragma unroll
    for (int c = 0; c < NC3; c++) {
      v16bf a = loadA_lds(c);
      v16bf bmNext = (c + 1 < NC3)
                   ? *(const v16bf*)(bst + (c+1)*FRAG_BYTES + lane*32) : bmCur;
      acc3 = __builtin_amdgcn_wmma_f32_16x16x32_bf16(
                 false, a, false, bmCur, (short)0, acc3, false, false);
      bmCur = bmNext;
    }
  }
  if (nloc < L_) {
    float bias3 = b3[nloc];
#pragma unroll
    for (int v = 0; v < 8; v++) {
      int rr = row0 + moff + v;
      out[(size_t)rr * L_ + nloc] = acc3[v] + bias3;
    }
  }
}

// ---------------------------------------------------------------------------
// Masked NLL over the 9-way logits, block-reduced, atomically accumulated.
// ---------------------------------------------------------------------------
__global__ __launch_bounds__(256)
void span_loss(const float* __restrict__ logits,
               const int* __restrict__ mask,
               const int* __restrict__ labels,
               float* __restrict__ loss) {
  __shared__ float red[256];
  int row = blockIdx.x * 256 + threadIdx.x;
  float val = 0.0f;
  if (row < ROWS && mask[row] == 1) {
    float x[L_];
    float m = -1e30f;
#pragma unroll
    for (int i = 0; i < L_; i++) { x[i] = logits[(size_t)row * L_ + i]; m = fmaxf(m, x[i]); }
    float s = 0.0f;
#pragma unroll
    for (int i = 0; i < L_; i++) s += expf(x[i] - m);
    float lse = m + logf(s);
    int lab = labels[row];
    lab = (lab < 0) ? 0 : ((lab > L_-1) ? L_-1 : lab);
    val = -(x[lab] - lse);
  }
  red[threadIdx.x] = val;
  __syncthreads();
  for (int off = 128; off > 0; off >>= 1) {
    if (threadIdx.x < (unsigned)off) red[threadIdx.x] += red[threadIdx.x + off];
    __syncthreads();
  }
  if (threadIdx.x == 0) atomicAdd(loss, red[0]);
}

extern "C" void kernel_launch(void* const* d_in, const int* in_sizes, int n_in,
                              void* d_out, int out_size, void* d_ws, size_t ws_size,
                              hipStream_t stream) {
  const float* seq    = (const float*)d_in[0];
  const float* wtab   = (const float*)d_in[1];
  const float* W1     = (const float*)d_in[2];
  const float* b1     = (const float*)d_in[3];
  const float* W2     = (const float*)d_in[4];
  const float* b2     = (const float*)d_in[5];
  const float* W3     = (const float*)d_in[6];
  const float* b3     = (const float*)d_in[7];
  const int*   spans  = (const int*)d_in[8];
  const int*   mask   = (const int*)d_in[9];
  const int*   labels = (const int*)d_in[10];
  float* out = (float*)d_out;
  char*  ws  = (char*)d_ws;

  float* loss_slot = out + (size_t)ROWS * L_;   /* logits occupy ROWS*9 floats */

  int prep_threads = NDTOT + WTB_DW;
  prep_swizzle<<<(prep_threads + 255) / 256, 256, 0, stream>>>(
      W1, W2, W3, wtab, (unsigned*)ws, loss_slot);
  prep_seq<<<(int)((SEQ_ELEMS/4 + 255) / 256), 256, 0, stream>>>(
      seq, (unsigned*)(ws + SEQB_OFF));
  span_mlp_wmma<<<ROWS / 128, 256, 0, stream>>>(b1, b2, b3, spans,
                                                (const char*)ws, out);
  span_loss<<<ROWS / 256, 256, 0, stream>>>(out, mask, labels, loss_slot);
}
